// MaxRetrievalModel_8340826489169
// MI455X (gfx1250) — compile-verified, matrix operand order checked by reading, not run on hardware
//
#include <hip/hip_runtime.h>
#include <hip/hip_bf16.h>
#include <math.h>

// ---------------------------------------------------------------------------
// Types for CDNA5 WMMA (wave32): A/B = 16 bf16 per lane, C/D = 8 f32 per lane
// ---------------------------------------------------------------------------
typedef __bf16 bf16;
typedef __attribute__((ext_vector_type(16))) __bf16 v16bf;
typedef __attribute__((ext_vector_type(8)))  __bf16 v8bf;
typedef __attribute__((ext_vector_type(8)))  float  v8f;

#define NB       32
#define NITEMS   8192
#define DITEM    128
#define DEMB     512
#define NCLASS   1000
#define TM       128          // item rows per block in the fused kernel

// Branch-free GELU (tanh form): gelu(x) = x * sigmoid(2u), u = x(c1 + c2 x^2).
// Lowers to v_exp_f32 + v_rcp_f32 + a few FMAs; no exec-mask divergence.
__device__ __forceinline__ float gelu_fast(float x) {
    float u = x * (0.7978845608028654f + 0.0356774081f * x * x);
    float e = __expf(-2.0f * u);
    return x * __builtin_amdgcn_rcpf(1.0f + e);
}

// A fragment: 16x32 bf16, row-major source (stride in elements).
// lanes 0-15: row M=lane, K=[0..7] then [16..23]; lanes 16-31: K=[8..15],[24..31].
__device__ __forceinline__ v16bf load_frag_a(const bf16* src, int row0, int stride, int k0) {
    int lane = threadIdx.x & 31;
    int m    = lane & 15;
    int khi  = (lane >> 4) << 3;
    const bf16* p = src + (size_t)(row0 + m) * stride + k0 + khi;
    v8bf lo = *(const v8bf*)p;
    v8bf hi = *(const v8bf*)(p + 16);
    v16bf a;
#pragma unroll
    for (int i = 0; i < 8; ++i) { a[i] = lo[i]; a[i + 8] = hi[i]; }
    return a;
}

// B fragment: 32x16 (KxN) where B[k][n] = w[n0+n][k], w row-major [nmax][din].
__device__ __forceinline__ v16bf load_frag_b(const bf16* w, int n0, int din, int k0, int nmax) {
    int lane = threadIdx.x & 31;
    int n    = lane & 15;
    int khi  = (lane >> 4) << 3;
    int row  = n0 + n; if (row >= nmax) row = nmax - 1;    // phi2 tail guard
    const bf16* p = w + (size_t)row * din + k0 + khi;
    v8bf lo = *(const v8bf*)p;
    v8bf hi = *(const v8bf*)(p + 16);
    v16bf b;
#pragma unroll
    for (int i = 0; i < 8; ++i) { b[i] = lo[i]; b[i + 8] = hi[i]; }
    return b;
}

__device__ __forceinline__ v8f wmma_bf16(v16bf a, v16bf b, v8f c) {
    return __builtin_amdgcn_wmma_f32_16x16x32_bf16(false, a, false, b, (short)0, c, false, false);
}

// C/D layout: VGPR r -> M = (lane>=16 ? 8 : 0) + r, N = lane & 15
__device__ __forceinline__ void store_tile_bf16(bf16* dst, int row0, int stride, int col0,
                                                const v8f& acc, bool do_gelu) {
    int lane = threadIdx.x & 31;
    int n    = lane & 15;
    int mhi  = (lane >> 4) << 3;
#pragma unroll
    for (int r = 0; r < 8; ++r) {
        float v = acc[r];
        if (do_gelu) v = gelu_fast(v);
        dst[(size_t)(row0 + mhi + r) * stride + col0 + n] = (bf16)v;
    }
}

// Software-pipelined 16x16xK GEMM tile over 8 row strips (K = KSTEPS*32, even).
// Ping-pong fragment buffers a0/b0 <-> a1/b1: loads of one stage are issued
// before the WMMA burst of the other, so 16 ds_load_b128 + 2 global_load_b128
// stay in flight behind every 8-WMMA burst. Buffers are overwritten, never
// copied, so the rolled loop carries no register moves.
template<int KSTEPS>
__device__ __forceinline__ void gemm_tile(const bf16* hsrc, int stride,
                                          const bf16* w, int n0, int din, int nmax,
                                          v8f acc[8]) {
    v16bf a0[8], a1[8], b0, b1;
    b0 = load_frag_b(w, n0, din, 0, nmax);
#pragma unroll
    for (int s = 0; s < 8; ++s) a0[s] = load_frag_a(hsrc, s * 16, stride, 0);

#pragma unroll 1
    for (int kc = 0; kc < KSTEPS; kc += 2) {
        b1 = load_frag_b(w, n0, din, (kc + 1) * 32, nmax);
#pragma unroll
        for (int s = 0; s < 8; ++s) a1[s] = load_frag_a(hsrc, s * 16, stride, (kc + 1) * 32);
#pragma unroll
        for (int s = 0; s < 8; ++s) acc[s] = wmma_bf16(a0[s], b0, acc[s]);

        if (kc + 2 < KSTEPS) {
            b0 = load_frag_b(w, n0, din, (kc + 2) * 32, nmax);
#pragma unroll
            for (int s = 0; s < 8; ++s) a0[s] = load_frag_a(hsrc, s * 16, stride, (kc + 2) * 32);
        }
#pragma unroll
        for (int s = 0; s < 8; ++s) acc[s] = wmma_bf16(a1[s], b1, acc[s]);
    }
}

// ---------------------------------------------------------------------------
// Utility kernels
// ---------------------------------------------------------------------------
__global__ void cvt_f32_bf16_kernel(const float* __restrict__ src, bf16* __restrict__ dst, int n) {
    for (int i = blockIdx.x * blockDim.x + threadIdx.x; i < n; i += gridDim.x * blockDim.x)
        dst[i] = (bf16)src[i];
}

__global__ void zero_f32_kernel(float* __restrict__ p, int n) {
    for (int i = blockIdx.x * blockDim.x + threadIdx.x; i < n; i += gridDim.x * blockDim.x)
        p[i] = 0.0f;
}

// ---------------------------------------------------------------------------
// Query path (tiny): qv[b][512] = q_w @ MLP(x_query[b]) + q_b
// ---------------------------------------------------------------------------
__global__ void query_kernel(const float* __restrict__ xq,
                             const float* __restrict__ w1, const float* __restrict__ b1,
                             const float* __restrict__ w2, const float* __restrict__ b2,
                             const float* __restrict__ qw, const float* __restrict__ qb,
                             float* __restrict__ qv) {
    __shared__ float h1[DEMB];
    __shared__ float h2[DEMB];
    int b = blockIdx.x;
    float x = xq[b];
    for (int j = threadIdx.x; j < DEMB; j += blockDim.x)
        h1[j] = gelu_fast(w1[j] * x + b1[j]);          // psi_q_w1 is [512,1]
    __syncthreads();
    for (int j = threadIdx.x; j < DEMB; j += blockDim.x) {
        float s = b2[j];
        const float* wr = w2 + (size_t)j * DEMB;
        for (int d = 0; d < DEMB; ++d) s += wr[d] * h1[d];
        h2[j] = s;
    }
    __syncthreads();
    for (int j = threadIdx.x; j < DEMB; j += blockDim.x) {
        float s = qb[j];
        const float* wr = qw + (size_t)j * DEMB;
        for (int d = 0; d < DEMB; ++d) s += wr[d] * h2[d];
        qv[(size_t)b * DEMB + j] = s;
    }
}

// ---------------------------------------------------------------------------
// Main fused kernel. Block = 256 threads (8 waves), TM=128 item rows.
// Wave w owns disjoint columns [w*64, w*64+64) over ALL 8 row strips:
//   -> every weight B-fragment is fetched exactly once per block (min L2 traffic)
//   -> v-phase columns are wave-private (direct global atomics, no LDS buffer)
// Occupancy is LDS-bound (289 KB -> 1 block/WGP, 2 waves/SIMD), so
// __launch_bounds__(256,1) frees the register budget for the pipeline buffers.
// ---------------------------------------------------------------------------
__global__ void __launch_bounds__(256, 1)
fused_items_kernel(const float* __restrict__ x_items,
                   const bf16*  __restrict__ w1b, const float* __restrict__ b1,
                   const bf16*  __restrict__ w2b, const float* __restrict__ b2,
                   const bf16*  __restrict__ kwb, const float* __restrict__ kb,
                   const bf16*  __restrict__ vwb, const float* __restrict__ vb,
                   const float* __restrict__ qv,
                   float* __restrict__ denom, float* __restrict__ zacc) {
    extern __shared__ char smem[];
    bf16*  xs = (bf16*)smem;               // [TM][128]  32 KB
    bf16*  hA = xs + TM * DITEM;           // [TM][512] 128 KB (h1, post-GELU)
    bf16*  hB = hA + TM * DEMB;            // [TM][512] 128 KB (h2 = h_items)
    float* ss = (float*)(hB + TM * DEMB);  // [TM] raw scores
    float* ps = ss + TM;                   // [TM] softmax numerators

    const int b    = blockIdx.y;
    const int tid  = threadIdx.x;
    const int wid  = tid >> 5;             // 0..7: column group
    const int lane = tid & 31;
    const int n    = lane & 15;
    const int mhi  = (lane >> 4) << 3;
    const int cbase = wid * 64;

    // ---- stage x tile (f32 -> bf16 LDS, packed 8-byte DS stores) ----
    const float* xsrc = x_items + ((size_t)b * NITEMS + (size_t)blockIdx.x * TM) * DITEM;
    for (int i = tid; i < TM * DITEM / 4; i += blockDim.x) {
        float4 f = ((const float4*)xsrc)[i];
        union { bf16 h[4]; uint2 u; } pk;
        pk.h[0] = (bf16)f.x; pk.h[1] = (bf16)f.y; pk.h[2] = (bf16)f.z; pk.h[3] = (bf16)f.w;
        *(uint2*)(xs + i * 4) = pk.u;
    }
    __syncthreads();

    // ---- layer 1: h1 = GELU(x @ w1^T + b1), K = 128 ----
    for (int ct = 0; ct < 4; ++ct) {
        int col0 = cbase + ct * 16;
        float bv = b1[col0 + n];
        v8f acc[8];
#pragma unroll
        for (int s = 0; s < 8; ++s) { v8f t = {bv,bv,bv,bv,bv,bv,bv,bv}; acc[s] = t; }
        gemm_tile<4>(xs, DITEM, w1b, col0, DITEM, DEMB, acc);
#pragma unroll
        for (int s = 0; s < 8; ++s) store_tile_bf16(hA, s * 16, DEMB, col0, acc[s], true);
    }
    __syncthreads();

    // ---- layer 2: h2 = h1 @ w2^T + b2, K = 512 ----
    for (int ct = 0; ct < 4; ++ct) {
        int col0 = cbase + ct * 16;
        float bv = b2[col0 + n];
        v8f acc[8];
#pragma unroll
        for (int s = 0; s < 8; ++s) { v8f t = {bv,bv,bv,bv,bv,bv,bv,bv}; acc[s] = t; }
        gemm_tile<16>(hA, DEMB, w2b, col0, DEMB, DEMB, acc);
#pragma unroll
        for (int s = 0; s < 8; ++s) store_tile_bf16(hB, s * 16, DEMB, col0, acc[s], false);
    }
    if (tid < TM) ss[tid] = 0.0f;
    __syncthreads();

    // ---- scores: s[m] = sum_d (h2 @ k_w^T + k_b)[m][d] * q[d] ----
    const float* qvb = qv + (size_t)b * DEMB;
    {
        float sc[8][8];
#pragma unroll
        for (int s = 0; s < 8; ++s)
#pragma unroll
            for (int r = 0; r < 8; ++r) sc[s][r] = 0.0f;

        for (int ct = 0; ct < 4; ++ct) {
            int col0 = cbase + ct * 16;
            float bv = kb[col0 + n];
            v8f acc[8];
#pragma unroll
            for (int s = 0; s < 8; ++s) { v8f t = {bv,bv,bv,bv,bv,bv,bv,bv}; acc[s] = t; }
            gemm_tile<16>(hB, DEMB, kwb, col0, DEMB, DEMB, acc);
            float qn = qvb[col0 + n];
#pragma unroll
            for (int s = 0; s < 8; ++s)
#pragma unroll
                for (int r = 0; r < 8; ++r) sc[s][r] += acc[s][r] * qn;
        }
#pragma unroll
        for (int s = 0; s < 8; ++s)
#pragma unroll
            for (int r = 0; r < 8; ++r) {
                float v = sc[s][r];
                v += __shfl_xor(v, 1, 16);
                v += __shfl_xor(v, 2, 16);
                v += __shfl_xor(v, 4, 16);
                v += __shfl_xor(v, 8, 16);
                sc[s][r] = v;
            }
        if (n == 0) {
#pragma unroll
            for (int s = 0; s < 8; ++s)
#pragma unroll
                for (int r = 0; r < 8; ++r)
                    atomicAdd(&ss[s * 16 + mhi + r], sc[s][r]);
        }
    }
    __syncthreads();

    if (tid < TM) ps[tid] = __expf(ss[tid] * 0.04419417382415922f);   // 1/sqrt(512)
    __syncthreads();
    if (tid < 32) {                         // one-wave parallel denominator reduce
        float dsum = ps[tid] + ps[tid + 32] + ps[tid + 64] + ps[tid + 96];
        dsum += __shfl_xor(dsum, 1);
        dsum += __shfl_xor(dsum, 2);
        dsum += __shfl_xor(dsum, 4);
        dsum += __shfl_xor(dsum, 8);
        dsum += __shfl_xor(dsum, 16);
        if (tid == 0) atomicAdd(&denom[b], dsum);
    }

    // ---- z partials: z[d] += sum_m p[m] * (h2 @ v_w^T + v_b)[m][d] ----
    for (int ct = 0; ct < 4; ++ct) {
        int col0 = cbase + ct * 16;
        float bv = vb[col0 + n];
        v8f acc[8];
#pragma unroll
        for (int s = 0; s < 8; ++s) { v8f t = {bv,bv,bv,bv,bv,bv,bv,bv}; acc[s] = t; }
        gemm_tile<16>(hB, DEMB, vwb, col0, DEMB, DEMB, acc);
        float zp = 0.0f;
#pragma unroll
        for (int s = 0; s < 8; ++s)
#pragma unroll
            for (int r = 0; r < 8; ++r) zp += ps[s * 16 + mhi + r] * acc[s][r];
        zp += __shfl_xor(zp, 16, 32);
        if (lane < 16) atomicAdd(&zacc[(size_t)b * DEMB + col0 + n], zp);   // wave-private column
    }
}

// ---------------------------------------------------------------------------
// Finalize: zbf = (zacc / denom) in bf16
// ---------------------------------------------------------------------------
__global__ void finalize_kernel(const float* __restrict__ zacc, const float* __restrict__ denom,
                                bf16* __restrict__ zbf) {
    int i = blockIdx.x * blockDim.x + threadIdx.x;
    if (i < NB * DEMB) zbf[i] = (bf16)(zacc[i] * __builtin_amdgcn_rcpf(denom[i >> 9]));
}

// ---------------------------------------------------------------------------
// Head: phi1 (GELU) then phi2 -> f32 logits. Block = 64 thr (2 row strips of
// the M=32 batch dim), grid = column tiles.
// ---------------------------------------------------------------------------
__global__ void phi1_kernel(const bf16* __restrict__ z, const bf16* __restrict__ w,
                            const float* __restrict__ bias, bf16* __restrict__ hout) {
    int col0 = blockIdx.x * 16;
    int row0 = (threadIdx.x >> 5) * 16;
    int lane = threadIdx.x & 31;
    float bv = bias[col0 + (lane & 15)];
    v8f acc = { bv, bv, bv, bv, bv, bv, bv, bv };
#pragma unroll
    for (int kc = 0; kc < 16; ++kc) {
        v16bf a  = load_frag_a(z, row0, DEMB, kc * 32);
        v16bf bb = load_frag_b(w, col0, DEMB, kc * 32, DEMB);
        acc = wmma_bf16(a, bb, acc);
    }
    store_tile_bf16(hout, row0, DEMB, col0, acc, true);
}

__global__ void phi2_kernel(const bf16* __restrict__ h, const bf16* __restrict__ w,
                            const float* __restrict__ bias, float* __restrict__ out) {
    int col0 = blockIdx.x * 16;
    int row0 = (threadIdx.x >> 5) * 16;
    int lane = threadIdx.x & 31;
    int n    = lane & 15;
    int mhi  = (lane >> 4) << 3;
    int colg = col0 + n;
    float bv = bias[colg < NCLASS ? colg : NCLASS - 1];
    v8f acc = { bv, bv, bv, bv, bv, bv, bv, bv };
#pragma unroll
    for (int kc = 0; kc < 16; ++kc) {
        v16bf a  = load_frag_a(h, row0, DEMB, kc * 32);
        v16bf bb = load_frag_b(w, col0, DEMB, kc * 32, NCLASS);
        acc = wmma_bf16(a, bb, acc);
    }
    if (colg < NCLASS) {
#pragma unroll
        for (int r = 0; r < 8; ++r)
            out[(size_t)(row0 + mhi + r) * NCLASS + colg] = acc[r];
    }
}

// ---------------------------------------------------------------------------
// Host launcher
// ---------------------------------------------------------------------------
extern "C" void kernel_launch(void* const* d_in, const int* in_sizes, int n_in,
                              void* d_out, int out_size, void* d_ws, size_t ws_size,
                              hipStream_t stream) {
    (void)in_sizes; (void)n_in; (void)out_size; (void)ws_size;

    const float* x_items  = (const float*)d_in[0];
    const float* x_query  = (const float*)d_in[1];
    const float* psi_x_w1 = (const float*)d_in[2];
    const float* psi_x_b1 = (const float*)d_in[3];
    const float* psi_x_w2 = (const float*)d_in[4];
    const float* psi_x_b2 = (const float*)d_in[5];
    const float* psi_q_w1 = (const float*)d_in[6];
    const float* psi_q_b1 = (const float*)d_in[7];
    const float* psi_q_w2 = (const float*)d_in[8];
    const float* psi_q_b2 = (const float*)d_in[9];
    const float* q_w      = (const float*)d_in[10];
    const float* q_b      = (const float*)d_in[11];
    const float* k_w      = (const float*)d_in[12];
    const float* k_b      = (const float*)d_in[13];
    const float* v_w      = (const float*)d_in[14];
    const float* v_b      = (const float*)d_in[15];
    const float* phi_w1   = (const float*)d_in[16];
    const float* phi_b1   = (const float*)d_in[17];
    const float* phi_w2   = (const float*)d_in[18];
    const float* phi_b2   = (const float*)d_in[19];
    float* out = (float*)d_out;

    // workspace carve-up (256B aligned)
    size_t off = 0;
    auto carve = [&](size_t bytes) -> char* {
        char* p = (char*)d_ws + off;
        off = (off + bytes + 255) & ~(size_t)255;
        return p;
    };
    float* qv    = (float*)carve(NB * DEMB * 4);
    float* denom = (float*)carve(NB * 4);
    float* zacc  = (float*)carve(NB * DEMB * 4);
    bf16*  zbf   = (bf16*) carve(NB * DEMB * 2);
    bf16*  hbf   = (bf16*) carve(NB * DEMB * 2);
    bf16*  w1b   = (bf16*) carve((size_t)DEMB * DITEM * 2);
    bf16*  w2b   = (bf16*) carve((size_t)DEMB * DEMB * 2);
    bf16*  kwb   = (bf16*) carve((size_t)DEMB * DEMB * 2);
    bf16*  vwb   = (bf16*) carve((size_t)DEMB * DEMB * 2);
    bf16*  p1b   = (bf16*) carve((size_t)DEMB * DEMB * 2);
    bf16*  p2b   = (bf16*) carve((size_t)NCLASS * DEMB * 2);

    // accumulators must be zeroed every call
    zero_f32_kernel<<<1, 64, 0, stream>>>(denom, NB);
    zero_f32_kernel<<<64, 256, 0, stream>>>(zacc, NB * DEMB);

    // weights -> bf16 (L2-resident working set, ~3.3 MB total)
    cvt_f32_bf16_kernel<<<256, 256, 0, stream>>>(psi_x_w1, w1b, DEMB * DITEM);
    cvt_f32_bf16_kernel<<<512, 256, 0, stream>>>(psi_x_w2, w2b, DEMB * DEMB);
    cvt_f32_bf16_kernel<<<512, 256, 0, stream>>>(k_w,      kwb, DEMB * DEMB);
    cvt_f32_bf16_kernel<<<512, 256, 0, stream>>>(v_w,      vwb, DEMB * DEMB);
    cvt_f32_bf16_kernel<<<512, 256, 0, stream>>>(phi_w1,   p1b, DEMB * DEMB);
    cvt_f32_bf16_kernel<<<512, 256, 0, stream>>>(phi_w2,   p2b, NCLASS * DEMB);

    // query path
    query_kernel<<<NB, 256, 0, stream>>>(x_query, psi_q_w1, psi_q_b1,
                                         psi_q_w2, psi_q_b2, q_w, q_b, qv);

    // main fused pass: ~289 KB dynamic LDS per block (1 block / 320 KB WGP)
    const size_t smem = (TM * DITEM + 2 * TM * DEMB) * sizeof(bf16)
                      + (2 * TM) * sizeof(float);
    hipFuncSetAttribute((const void*)fused_items_kernel,
                        hipFuncAttributeMaxDynamicSharedMemorySize, (int)smem);
    fused_items_kernel<<<dim3(NITEMS / TM, NB), 256, smem, stream>>>(
        x_items, w1b, psi_x_b1, w2b, psi_x_b2,
        kwb, k_b, vwb, v_b, qv, denom, zacc);

    // finalize + head
    finalize_kernel<<<(NB * DEMB + 255) / 256, 256, 0, stream>>>(zacc, denom, zbf);
    phi1_kernel<<<DEMB / 16, 64, 0, stream>>>(zbf, p1b, phi_b1, hbf);
    phi2_kernel<<<(NCLASS + 15) / 16, 64, 0, stream>>>(hbf, p2b, phi_b2, out);
}
